// Net_20925080666587
// MI455X (gfx1250) — matrix operand order repaired and checked by reading, not compiled, not run on hardware
//
#include <hip/hip_runtime.h>
#include <hip/hip_bf16.h>

#define NNODES 65536
#define NEDGES 1048576
#define NGRAPH 64
#define NPG    1024
#define HDIM   128
#define KKEEP  820

typedef __attribute__((ext_vector_type(16))) __bf16 v16bf;
typedef __attribute__((ext_vector_type(8)))  float  v8f;

// ---------------------------------------------------------------------------
// Weight prep: wT[n*256 + k] = bf16( k<128 ? Wrel[k][n] : Wroot[k-128][n] )
// ---------------------------------------------------------------------------
__global__ void prep_w_kernel(const float* __restrict__ Wrel,
                              const float* __restrict__ Wroot,
                              __bf16* __restrict__ wT) {
    int idx = blockIdx.x * blockDim.x + threadIdx.x;   // 128*256
    if (idx >= HDIM * 256) return;
    int n = idx >> 8;
    int k = idx & 255;
    float v = (k < HDIM) ? Wrel[k * HDIM + n] : Wroot[(k - HDIM) * HDIM + n];
    wT[n * 256 + k] = (__bf16)v;
}

// ---------------------------------------------------------------------------
// Edge scatter (mean-agg numerator + degree). One wave per edge, 4 feats/lane.
// mode 0: unmasked, accumulate deg. mode 1: unmasked, no deg.
// mode 2: masked by nmask on both endpoints, accumulate deg.
// ---------------------------------------------------------------------------
__global__ void scatter_kernel(const float* __restrict__ x,
                               const int* __restrict__ src,
                               const int* __restrict__ dst,
                               float* __restrict__ agg,
                               float* __restrict__ deg,
                               const float* __restrict__ nmask,
                               int mode) {
    int e    = (blockIdx.x * blockDim.x + threadIdx.x) >> 5;
    int lane = threadIdx.x & 31;
    if (e >= NEDGES) return;
    int s = src[e];
    int d = dst[e];
    if (mode == 2) {
        if (nmask[s] < 0.5f || nmask[d] < 0.5f) return;
    }
    const float4 v = *(const float4*)(x + (size_t)s * HDIM + lane * 4);
    float* ap = agg + (size_t)d * HDIM + lane * 4;
    atomicAdd(ap + 0, v.x);
    atomicAdd(ap + 1, v.y);
    atomicAdd(ap + 2, v.z);
    atomicAdd(ap + 3, v.w);
    if (mode != 1 && lane == 0) atomicAdd(deg + d, 1.0f);
}

// ---------------------------------------------------------------------------
// Fused GraphConv GEMM: out = relu( (agg/max(deg,1)) @ Wrel + b + xroot @ Wroot )
// K=256 concat GEMM via v_wmma_f32_16x16x32_bf16.
// Block = 256 thr (8 waves), block tile = 128 rows x 128 cols,
// wave tile = 16 rows x 128 cols = 8 ntiles, K loop = 8 ktiles of 32.
// ---------------------------------------------------------------------------
__global__ __launch_bounds__(256) void gconv_gemm_kernel(
        const float* __restrict__ agg,
        const float* __restrict__ xroot,
        const float* __restrict__ deg,
        const __bf16* __restrict__ wT,     // [128 out][256 k] bf16
        const float* __restrict__ bias,
        float* __restrict__ out) {
    const int wave = threadIdx.x >> 5;
    const int lane = threadIdx.x & 31;
    const int lm   = lane & 15;
    const int lh   = lane >> 4;            // 0 or 1
    const int m0   = blockIdx.x * 128 + wave * 16;
    const int row  = m0 + lm;

    const float sc = 1.0f / fmaxf(deg[row], 1.0f);

    v8f acc[8];
#pragma unroll
    for (int nt = 0; nt < 8; ++nt)
        acc[nt] = (v8f){0.f, 0.f, 0.f, 0.f, 0.f, 0.f, 0.f, 0.f};

#pragma unroll
    for (int kt = 0; kt < 8; ++kt) {
        const float* sp = ((kt < 4) ? agg : xroot)
                          + (size_t)row * HDIM + (kt & 3) * 32 + lh * 16;
        float4 p0 = *(const float4*)(sp + 0);
        float4 p1 = *(const float4*)(sp + 4);
        float4 p2 = *(const float4*)(sp + 8);
        float4 p3 = *(const float4*)(sp + 12);
        if (kt < 4) {
            p0.x *= sc; p0.y *= sc; p0.z *= sc; p0.w *= sc;
            p1.x *= sc; p1.y *= sc; p1.z *= sc; p1.w *= sc;
            p2.x *= sc; p2.y *= sc; p2.z *= sc; p2.w *= sc;
            p3.x *= sc; p3.y *= sc; p3.z *= sc; p3.w *= sc;
        }
        v16bf a;
        a[0]  = (__bf16)p0.x; a[1]  = (__bf16)p0.y;
        a[2]  = (__bf16)p0.z; a[3]  = (__bf16)p0.w;
        a[4]  = (__bf16)p1.x; a[5]  = (__bf16)p1.y;
        a[6]  = (__bf16)p1.z; a[7]  = (__bf16)p1.w;
        a[8]  = (__bf16)p2.x; a[9]  = (__bf16)p2.y;
        a[10] = (__bf16)p2.z; a[11] = (__bf16)p2.w;
        a[12] = (__bf16)p3.x; a[13] = (__bf16)p3.y;
        a[14] = (__bf16)p3.z; a[15] = (__bf16)p3.w;

#pragma unroll
        for (int nt = 0; nt < 8; ++nt) {
            const __bf16* bp = wT + (size_t)(nt * 16 + lm) * 256 + kt * 32 + lh * 16;
            v16bf b = *(const v16bf*)bp;
            acc[nt] = __builtin_amdgcn_wmma_f32_16x16x32_bf16(
                false, a, false, b, (short)0, acc[nt], false, false);
        }
    }

    // epilogue: C/D layout -> VGPR v: lanes 0-15 row m0+v, lanes 16-31 row m0+8+v; col = lane%16
#pragma unroll
    for (int nt = 0; nt < 8; ++nt) {
        const int col = nt * 16 + lm;
        const float bb = bias[col];
#pragma unroll
        for (int v = 0; v < 8; ++v) {
            const int r = m0 + v + lh * 8;
            out[(size_t)r * HDIM + col] = fmaxf(acc[nt][v] + bb, 0.0f);
        }
    }
}

// ---------------------------------------------------------------------------
// SAGPool score: GraphConv(H -> 1). One wave per node.
// ---------------------------------------------------------------------------
__global__ void score_kernel(const float* __restrict__ agg,
                             const float* __restrict__ deg,
                             const float* __restrict__ x,
                             const float* __restrict__ Wsrel,
                             const float* __restrict__ bs,
                             const float* __restrict__ Wsroot,
                             float* __restrict__ score) {
    int n    = (blockIdx.x * blockDim.x + threadIdx.x) >> 5;
    int lane = threadIdx.x & 31;
    if (n >= NNODES) return;
    float rdeg = 1.0f / fmaxf(deg[n], 1.0f);
    float part = 0.0f;
#pragma unroll
    for (int i = 0; i < 4; ++i) {
        int f = lane * 4 + i;
        part += agg[(size_t)n * HDIM + f] * rdeg * Wsrel[f];
        part += x[(size_t)n * HDIM + f] * Wsroot[f];
    }
#pragma unroll
    for (int off = 16; off > 0; off >>= 1)
        part += __shfl_xor(part, off, 32);
    if (lane == 0) score[n] = part + bs[0];
}

// ---------------------------------------------------------------------------
// Per-graph exact top-K via bitonic sort of (sortable_score, ~idx) u64 keys.
// 64 blocks x 512 threads; 8KB LDS. Writes full nmask (0/1) for the graph.
// ---------------------------------------------------------------------------
__global__ __launch_bounds__(512) void topk_kernel(const float* __restrict__ score,
                                                   float* __restrict__ nmask) {
    __shared__ unsigned long long keys[NPG];
    const int g = blockIdx.x;
    const int t = threadIdx.x;
    for (int i = t; i < NPG; i += 512) {
        unsigned u = __float_as_uint(score[g * NPG + i]);
        u = (u & 0x80000000u) ? ~u : (u | 0x80000000u);   // monotonic map
        keys[i] = ((unsigned long long)u << 32) | (unsigned)(0xFFFFFFFFu - (unsigned)i);
    }
    __syncthreads();
    for (int k = 2; k <= NPG; k <<= 1) {
        for (int j = k >> 1; j > 0; j >>= 1) {
            for (int i = t; i < NPG; i += 512) {
                int ixj = i ^ j;
                if (ixj > i) {
                    bool up = ((i & k) == 0);               // ascending segment
                    unsigned long long a = keys[i], b = keys[ixj];
                    bool sw = up ? (a > b) : (a < b);
                    if (sw) { keys[i] = b; keys[ixj] = a; }
                }
            }
            __syncthreads();
        }
    }
    // ascending sort: positions [NPG-K, NPG) are the kept nodes
    for (int i = t; i < NPG; i += 512) {
        unsigned idx = 0xFFFFFFFFu - (unsigned)(keys[i] & 0xFFFFFFFFull);
        nmask[g * NPG + idx] = (i >= NPG - KKEEP) ? 1.0f : 0.0f;
    }
}

// ---------------------------------------------------------------------------
// Gate: x *= tanh(score[n]) * nmask[n]   (in place over [N,H])
// ---------------------------------------------------------------------------
__global__ void gate_kernel(float* __restrict__ x,
                            const float* __restrict__ score,
                            const float* __restrict__ nmask) {
    size_t idx = (size_t)blockIdx.x * blockDim.x + threadIdx.x;
    if (idx >= (size_t)NNODES * HDIM) return;
    int n = (int)(idx >> 7);
    x[idx] *= tanhf(score[n]) * nmask[n];
}

// ---------------------------------------------------------------------------
// Global mean pool per graph. One block (128 thr) per graph.
// Writes pooled[g*384 + t] (caller offsets base pointer by stage*128).
// ---------------------------------------------------------------------------
__global__ void pool_kernel(const float* __restrict__ x,
                            const float* __restrict__ nmask,   // may be null
                            float inv_cnt,
                            float* __restrict__ outp) {
    const int g = blockIdx.x;
    const int t = threadIdx.x;
    float s = 0.0f;
    for (int i = 0; i < NPG; ++i) {
        int n = g * NPG + i;
        float m = nmask ? nmask[n] : 1.0f;
        s += x[(size_t)n * HDIM + t] * m;
    }
    outp[g * 384 + t] = s * inv_cnt;
}

// ---------------------------------------------------------------------------
// MLP head: relu([B,384]@[384,128]+b) @ [128,10]+b -> log_softmax
// One block (128 thr) per graph.
// ---------------------------------------------------------------------------
__global__ void head_kernel(const float* __restrict__ pooled,
                            const float* __restrict__ Wl1,
                            const float* __restrict__ bl1,
                            const float* __restrict__ Wl2,
                            const float* __restrict__ bl2,
                            float* __restrict__ out) {
    __shared__ float cat[384];
    __shared__ float h1[HDIM];
    __shared__ float lg[16];
    __shared__ float red[2];
    const int g = blockIdx.x;
    const int t = threadIdx.x;
    for (int i = t; i < 384; i += HDIM) cat[i] = pooled[g * 384 + i];
    __syncthreads();
    float acc = bl1[t];
    for (int k = 0; k < 384; ++k) acc += cat[k] * Wl1[k * HDIM + t];
    h1[t] = fmaxf(acc, 0.0f);
    __syncthreads();
    if (t < 10) {
        float a = bl2[t];
        for (int j = 0; j < HDIM; ++j) a += h1[j] * Wl2[j * 10 + t];
        lg[t] = a;
    }
    __syncthreads();
    if (t == 0) {
        float m = lg[0];
        for (int c = 1; c < 10; ++c) m = fmaxf(m, lg[c]);
        float s = 0.0f;
        for (int c = 0; c < 10; ++c) s += expf(lg[c] - m);
        red[0] = m;
        red[1] = logf(s);
    }
    __syncthreads();
    if (t < 10) out[g * 10 + t] = lg[t] - red[0] - red[1];
}

// ---------------------------------------------------------------------------
extern "C" void kernel_launch(void* const* d_in, const int* in_sizes, int n_in,
                              void* d_out, int out_size, void* d_ws, size_t ws_size,
                              hipStream_t stream) {
    (void)in_sizes; (void)n_in; (void)out_size; (void)ws_size;

    const float* x0     = (const float*)d_in[0];
    const int*   src    = (const int*)d_in[1];
    const int*   dst    = (const int*)d_in[2];
    const float* W1r    = (const float*)d_in[3];
    const float* b1     = (const float*)d_in[4];
    const float* W1o    = (const float*)d_in[5];
    const float* W2r    = (const float*)d_in[6];
    const float* b2     = (const float*)d_in[7];
    const float* W2o    = (const float*)d_in[8];
    const float* W3r    = (const float*)d_in[9];
    const float* b3     = (const float*)d_in[10];
    const float* W3o    = (const float*)d_in[11];
    const float* Wsr    = (const float*)d_in[12];
    const float* bs     = (const float*)d_in[13];
    const float* Wso    = (const float*)d_in[14];
    const float* Wl1    = (const float*)d_in[15];
    const float* bl1    = (const float*)d_in[16];
    const float* Wl2    = (const float*)d_in[17];
    const float* bl2    = (const float*)d_in[18];
    float* out = (float*)d_out;

    // workspace layout
    float* ws     = (float*)d_ws;
    float* agg    = ws;                                  // N*H
    float* bufA   = agg  + (size_t)NNODES * HDIM;        // x1 / x3
    float* bufB   = bufA + (size_t)NNODES * HDIM;        // x2 (gated in place)
    float* deg    = bufB + (size_t)NNODES * HDIM;        // N
    float* score  = deg   + NNODES;                      // N
    float* nmask  = score + NNODES;                      // N
    float* pooled = nmask + NNODES;                      // 64*384
    __bf16* wT1 = (__bf16*)(pooled + NGRAPH * 384);
    __bf16* wT2 = wT1 + HDIM * 256;
    __bf16* wT3 = wT2 + HDIM * 256;

    const size_t aggBytes = (size_t)NNODES * HDIM * sizeof(float);
    const int scatterBlocks = NEDGES / 8;                // wave per edge, 8 waves/block
    const int gemmBlocks    = NNODES / 128;

    // weight prep
    prep_w_kernel<<<128, 256, 0, stream>>>(W1r, W1o, wT1);
    prep_w_kernel<<<128, 256, 0, stream>>>(W2r, W2o, wT2);
    prep_w_kernel<<<128, 256, 0, stream>>>(W3r, W3o, wT3);

    // conv1: agg(x0), deg0 ; x1 = relu(gemm)
    hipMemsetAsync(agg, 0, aggBytes, stream);
    hipMemsetAsync(deg, 0, NNODES * sizeof(float), stream);
    scatter_kernel<<<scatterBlocks, 256, 0, stream>>>(x0, src, dst, agg, deg, nullptr, 0);
    gconv_gemm_kernel<<<gemmBlocks, 256, 0, stream>>>(agg, x0, deg, wT1, b1, bufA);
    pool_kernel<<<NGRAPH, HDIM, 0, stream>>>(bufA, nullptr, 1.0f / NPG, pooled + 0);

    // conv2: agg(x1) ; x2 = relu(gemm)
    hipMemsetAsync(agg, 0, aggBytes, stream);
    scatter_kernel<<<scatterBlocks, 256, 0, stream>>>(bufA, src, dst, agg, nullptr, nullptr, 1);
    gconv_gemm_kernel<<<gemmBlocks, 256, 0, stream>>>(agg, bufA, deg, wT2, b2, bufB);
    pool_kernel<<<NGRAPH, HDIM, 0, stream>>>(bufB, nullptr, 1.0f / NPG, pooled + 128);

    // SAGPool score: agg(x2) -> GraphConv(H,1)
    hipMemsetAsync(agg, 0, aggBytes, stream);
    scatter_kernel<<<scatterBlocks, 256, 0, stream>>>(bufB, src, dst, agg, nullptr, nullptr, 1);
    score_kernel<<<NNODES / 8, 256, 0, stream>>>(agg, deg, bufB, Wsr, bs, Wso, score);

    // top-K keep mask, then gate x2 in place
    topk_kernel<<<NGRAPH, 512, 0, stream>>>(score, nmask);
    gate_kernel<<<(NNODES * HDIM) / 256, 256, 0, stream>>>(bufB, score, nmask);

    // conv3 (masked edges): agg(x2g) with nmask on both endpoints, masked deg
    hipMemsetAsync(agg, 0, aggBytes, stream);
    hipMemsetAsync(deg, 0, NNODES * sizeof(float), stream);
    scatter_kernel<<<scatterBlocks, 256, 0, stream>>>(bufB, src, dst, agg, deg, nmask, 2);
    gconv_gemm_kernel<<<gemmBlocks, 256, 0, stream>>>(agg, bufB, deg, wT3, b3, bufA);
    pool_kernel<<<NGRAPH, HDIM, 0, stream>>>(bufA, nmask, 1.0f / KKEEP, pooled + 256);

    // MLP head + log_softmax
    head_kernel<<<NGRAPH, HDIM, 0, stream>>>(pooled, Wl1, bl1, Wl2, bl2, out);
}
